// HardnetLoss_49959059587631
// MI455X (gfx1250) — matrix operand order repaired and checked by reading, not compiled, not run on hardware
//
#include <hip/hip_runtime.h>
#include <hip/hip_bf16.h>

typedef __attribute__((ext_vector_type(16))) _Float16 v16h;
typedef __attribute__((ext_vector_type(8)))  _Float16 v8h;
typedef __attribute__((ext_vector_type(4)))  _Float16 v4h;
typedef __attribute__((ext_vector_type(8)))  float    v8f;

constexpr int Bn   = 8;
constexpr int Cc   = 128;
constexpr int Hh   = 192;
constexpr int Ww   = 192;
constexpr int LEN  = 39 / 2;        // 19
constexpr int S0   = 96 - LEN;      // 77
constexpr int SL   = 2 * LEN;       // 38
constexpr int NROW = Bn * SL * SL;  // 11552
constexpr int NT   = NROW / 16;     // 722 exact 16-tiles
constexpr int NBLK = NT / 2;        // 361 blocks, 2 row tiles each

// dm < 0.008  <=>  sqrt(sqc + 1e-6) + 1e-8 < 0.008  <=>  sqc < TSQ
constexpr float TSQ  = 6.29998400e-05f;   // (0.008 - 1e-8)^2 - 1e-6
constexpr float FINF = 3.0e38f;

// ---------------------------------------------------------------------------
// Kernel 1: gather f1 crop + bilinear-sample f2 at aflow coords; emit f16 rows
// ---------------------------------------------------------------------------
__device__ __forceinline__ float gsample(const float* __restrict__ f, float xf, float yf) {
    const bool valid = (xf >= 0.0f) && (xf <= (float)(Ww - 1)) &&
                       (yf >= 0.0f) && (yf <= (float)(Hh - 1));
    const int xi = (int)fminf(fmaxf(xf, 0.0f), (float)(Ww - 1));
    const int yi = (int)fminf(fmaxf(yf, 0.0f), (float)(Hh - 1));
    const float v = f[yi * Ww + xi];
    return valid ? v : 0.0f;
}

__global__ __launch_bounds__(128)
void prep_kernel(const float* __restrict__ feat1, const float* __restrict__ feat2,
                 const float* __restrict__ aflow,
                 _Float16* __restrict__ a16, _Float16* __restrict__ p16) {
    const int r  = blockIdx.x;          // row in [0, NROW)
    const int c  = threadIdx.x;         // channel
    const int ix = r % SL;
    const int iy = (r / SL) % SL;
    const int b  = r / (SL * SL);
    const int x  = S0 + ix;
    const int y  = S0 + iy;

    const float av = feat1[(((size_t)b * Cc + c) * Hh + y) * Ww + x];

    float gnx = aflow[(((size_t)b * 2 + 0) * Hh + y) * Ww + x] * (2.0f / (Ww - 1)) - 1.0f;
    float gny = aflow[(((size_t)b * 2 + 1) * Hh + y) * Ww + x] * (2.0f / (Hh - 1)) - 1.0f;
    if (gnx != gnx) gnx = 9.0e9f;       // NaN -> huge -> sampled as zeros
    if (gny != gny) gny = 9.0e9f;
    const float px = (gnx + 1.0f) * 0.5f * (float)(Ww - 1);
    const float py = (gny + 1.0f) * 0.5f * (float)(Hh - 1);
    const float x0 = floorf(px), y0 = floorf(py);
    const float wx1 = px - x0, wx0 = 1.0f - wx1;
    const float wy1 = py - y0, wy0 = 1.0f - wy1;

    const float* __restrict__ f2c = feat2 + ((size_t)b * Cc + c) * (size_t)(Hh * Ww);
    const float pv = wx0 * wy0 * gsample(f2c, x0,        y0)
                   + wx1 * wy0 * gsample(f2c, x0 + 1.0f, y0)
                   + wx0 * wy1 * gsample(f2c, x0,        y0 + 1.0f)
                   + wx1 * wy1 * gsample(f2c, x0 + 1.0f, y0 + 1.0f);

    a16[(size_t)r * Cc + c] = (_Float16)av;
    p16[(size_t)r * Cc + c] = (_Float16)pv;
}

// ---------------------------------------------------------------------------
// Kernel 2: per-row squared norms + diagonal distance (pos), all from the
//           f16-rounded values so the diag is consistent with the GEMM
// ---------------------------------------------------------------------------
__global__ __launch_bounds__(32)
void norms_kernel(const _Float16* __restrict__ a16, const _Float16* __restrict__ p16,
                  float* __restrict__ d1, float* __restrict__ d2,
                  float* __restrict__ pos) {
    const int r    = blockIdx.x;
    const int lane = threadIdx.x;
    const v4h va = *(const v4h*)(a16 + (size_t)r * Cc + lane * 4);
    const v4h vp = *(const v4h*)(p16 + (size_t)r * Cc + lane * 4);
    float sa = 0.0f, sp = 0.0f, sd = 0.0f;
#pragma unroll
    for (int i = 0; i < 4; ++i) {
        const float fa = (float)va[i], fp = (float)vp[i];
        sa += fa * fa; sp += fp * fp; sd += fa * fp;
    }
#pragma unroll
    for (int o = 16; o > 0; o >>= 1) {
        sa += __shfl_xor(sa, o, 32);
        sp += __shfl_xor(sp, o, 32);
        sd += __shfl_xor(sd, o, 32);
    }
    if (lane == 0) {
        d1[r] = sa;
        d2[r] = sp;
        const float sq = fmaxf(sa + sp - 2.0f * sd, 0.0f);
        pos[r] = __builtin_amdgcn_sqrtf(sq + 1e-6f) + 1e-8f;   // diagonal of dm
    }
}

// ---------------------------------------------------------------------------
// Kernel 3: fused WMMA GEMM + sq-domain row-min. Branchless epilogue: diag
//           exclusion via mask arithmetic (diag value handled analytically).
//           block = 4 waves owning TWO 16-row tiles; waves stride column
//           tiles; B tiles software double-buffered.
// ---------------------------------------------------------------------------
__device__ __forceinline__ void load_bfrag(const _Float16* __restrict__ p16,
                                           int j, int idx, int half, v16h bf[4]) {
    // B = p^T : lane holds column n = idx; per-lane K = kk*32 + half*8 + {0..7, 16..23}
    const _Float16* __restrict__ prow = p16 + (size_t)(j * 16 + idx) * Cc;
#pragma unroll
    for (int kk = 0; kk < 4; ++kk) {
        const int kb = kk * 32 + half * 8;
        const v8h lo = *(const v8h*)(prow + kb);
        const v8h hi = *(const v8h*)(prow + kb + 16);
        bf[kk] = __builtin_shufflevector(lo, hi, 0,1,2,3,4,5,6,7,8,9,10,11,12,13,14,15);
    }
}

__global__ __launch_bounds__(128)
void hardnet_gemm_kernel(const _Float16* __restrict__ a16, const _Float16* __restrict__ p16,
                         const float* __restrict__ d1, const float* __restrict__ d2,
                         const float* __restrict__ pos, float* __restrict__ minneg) {
    const int rt0  = blockIdx.x * 2;    // first of two row tiles
    const int wave = threadIdx.x >> 5;
    const int lane = threadIdx.x & 31;
    const int idx  = lane & 15;         // m for A, n for B/C/D
    const int half = lane >> 4;

    __shared__ float sminN[32][64];     // normal candidates (sq domain)
    __shared__ float sminM[32][64];     // masked (<0.008) candidates (sq domain)

    // A fragments for both row tiles, resident in VGPRs for the whole kernel.
    // 16-bit A 16x32 layout: lane holds row m=idx; K = kk*32 + half*8 + {0..7, 16..23}
    v16h afrag[2][4];
#pragma unroll
    for (int t = 0; t < 2; ++t) {
        const _Float16* __restrict__ arow = a16 + (size_t)((rt0 + t) * 16 + idx) * Cc;
#pragma unroll
        for (int kk = 0; kk < 4; ++kk) {
            const int kb = kk * 32 + half * 8;
            const v8h lo = *(const v8h*)(arow + kb);
            const v8h hi = *(const v8h*)(arow + kb + 16);
            afrag[t][kk] = __builtin_shufflevector(lo, hi, 0,1,2,3,4,5,6,7,8,9,10,11,12,13,14,15);
        }
    }
    float d1v[2][8], runN[2][8], runM[2][8];
    bool diagmask[8];                   // loop-invariant: lane owns row i+8*half?
#pragma unroll
    for (int i = 0; i < 8; ++i) diagmask[i] = (idx == i + 8 * half);
#pragma unroll
    for (int t = 0; t < 2; ++t)
#pragma unroll
        for (int i = 0; i < 8; ++i) {
            d1v[t][i] = d1[(rt0 + t) * 16 + i + 8 * half];
            runN[t][i] = FINF;
            runM[t][i] = FINF;
        }

    // Software-pipelined column-tile loop (double-buffered B fragments).
    v16h bcur[4];
    load_bfrag(p16, wave, idx, half, bcur);   // wave < 4 <= NT, always valid

    for (int j = wave; j < NT; j += 4) {
        v16h bnxt[4];
        if (j + 4 < NT) load_bfrag(p16, j + 4, idx, half, bnxt);
        const float d2v = d2[j * 16 + idx];

        v8f acc[2];
#pragma unroll
        for (int t = 0; t < 2; ++t) {
            acc[t] = v8f{};
#pragma unroll
            for (int kk = 0; kk < 4; ++kk)
                acc[t] = __builtin_amdgcn_wmma_f32_16x16x32_f16(
                    false, afrag[t][kk], false, bcur[kk], (short)0, acc[t], false, false);
        }

#pragma unroll
        for (int t = 0; t < 2; ++t) {
            const bool diagtile = (j == rt0 + t);
#pragma unroll
            for (int i = 0; i < 8; ++i) {
                // D layout: element i is row m = i + 8*half, col n = idx
                const float sqc = fmaxf(d1v[t][i] + d2v - 2.0f * acc[t][i], 0.0f);
                const bool isdiag = diagtile & diagmask[i];      // no control flow
                const bool masked = (sqc < TSQ);
                const float keyN = (masked | isdiag) ? FINF : sqc;
                const float keyM = (masked & !isdiag) ? sqc : FINF;
                runN[t][i] = fminf(runN[t][i], keyN);
                runM[t][i] = fminf(runM[t][i], keyM);
            }
        }
#pragma unroll
        for (int kk = 0; kk < 4; ++kk) bcur[kk] = bnxt[kk];
    }

    // Combine partial mins: row m of tile t lives in 16 lanes x 4 waves
#pragma unroll
    for (int t = 0; t < 2; ++t)
#pragma unroll
        for (int i = 0; i < 8; ++i) {
            sminN[t * 16 + i + 8 * half][wave * 16 + idx] = runN[t][i];
            sminM[t * 16 + i + 8 * half][wave * 16 + idx] = runM[t][i];
        }
    __syncthreads();
    if (threadIdx.x < 32) {
        float mN = FINF, mM = FINF;
        for (int k = 0; k < 64; ++k) {
            mN = fminf(mN, sminN[threadIdx.x][k]);
            mM = fminf(mM, sminM[threadIdx.x][k]);
        }
        // back to distance domain: one sqrt per candidate class per row
        const float dmN = __builtin_amdgcn_sqrtf(mN + 1e-6f) + 1e-8f;
        const float dmM = __builtin_amdgcn_sqrtf(mM + 1e-6f) + 1e-8f + 10.0f;
        const int row = rt0 * 16 + threadIdx.x;
        const float diagc = pos[row] + 10.0f;     // diag candidate, never re-masked
        minneg[row] = fminf(fminf(dmN, dmM), diagc);
    }
}

// ---------------------------------------------------------------------------
// Kernel 4: loss = mean(relu(margin + pos - min_neg))
// ---------------------------------------------------------------------------
__global__ __launch_bounds__(256)
void loss_kernel(const float* __restrict__ pos, const float* __restrict__ minneg,
                 float* __restrict__ out) {
    __shared__ float sdata[256];
    const int tid = threadIdx.x;
    float s = 0.0f;
    for (int i = tid; i < NROW; i += 256)
        s += fmaxf(1.0f + pos[i] - minneg[i], 0.0f);
    sdata[tid] = s;
    __syncthreads();
    for (int o = 128; o > 0; o >>= 1) {
        if (tid < o) sdata[tid] += sdata[tid + o];
        __syncthreads();
    }
    if (tid == 0) out[0] = sdata[0] * (1.0f / (float)NROW);
}

// ---------------------------------------------------------------------------
extern "C" void kernel_launch(void* const* d_in, const int* in_sizes, int n_in,
                              void* d_out, int out_size, void* d_ws, size_t ws_size,
                              hipStream_t stream) {
    (void)in_sizes; (void)n_in; (void)out_size; (void)ws_size;
    const float* feat1 = (const float*)d_in[0];
    const float* feat2 = (const float*)d_in[1];
    const float* aflow = (const float*)d_in[2];
    float* out = (float*)d_out;

    char* ws = (char*)d_ws;
    auto align256 = [](size_t x) { return (x + 255) & ~(size_t)255; };
    size_t off = 0;
    _Float16* a16 = (_Float16*)(ws + off); off = align256(off + (size_t)NROW * Cc * sizeof(_Float16));
    _Float16* p16 = (_Float16*)(ws + off); off = align256(off + (size_t)NROW * Cc * sizeof(_Float16));
    float* d1     = (float*)(ws + off);    off = align256(off + (size_t)NROW * sizeof(float));
    float* d2     = (float*)(ws + off);    off = align256(off + (size_t)NROW * sizeof(float));
    float* pos    = (float*)(ws + off);    off = align256(off + (size_t)NROW * sizeof(float));
    float* minneg = (float*)(ws + off);    off = align256(off + (size_t)NROW * sizeof(float));

    prep_kernel<<<NROW, 128, 0, stream>>>(feat1, feat2, aflow, a16, p16);
    norms_kernel<<<NROW, 32, 0, stream>>>(a16, p16, d1, d2, pos);
    hardnet_gemm_kernel<<<NBLK, 128, 0, stream>>>(a16, p16, d1, d2, pos, minneg);
    loss_kernel<<<1, 256, 0, stream>>>(pos, minneg, out);
}
